// MAB_37890201485318
// MI455X (gfx1250) — compile-verified, matrix operand order checked by reading, not run on hardware
//
#include <hip/hip_runtime.h>

typedef __attribute__((ext_vector_type(16))) __bf16 v16bf;
typedef __attribute__((ext_vector_type(8)))  __bf16 bf16x8;
typedef __attribute__((ext_vector_type(8)))  float  v8f;
typedef __attribute__((ext_vector_type(4)))  float  f32x4;

#define NEGBIG (-1.0e9f)

__device__ __forceinline__ v8f vzero8() {
  v8f z;
#pragma unroll
  for (int i = 0; i < 8; ++i) z[i] = 0.0f;
  return z;
}

__device__ __forceinline__ v16bf pack16(bf16x8 lo, bf16x8 hi) {
  return __builtin_shufflevector(lo, hi, 0, 1, 2, 3, 4, 5, 6, 7, 8, 9, 10, 11, 12, 13, 14, 15);
}

// Load a 16x32 bf16 WMMA A/B fragment from a row-major tile.
// A: row m = lane&15, K halves selected by lane>>4 (per ISA 16-bit A 16x32 layout).
// B is the symmetric layout with n = lane&15 reading rows of the N-major (transposed) tile.
__device__ __forceinline__ v16bf load_frag(const __bf16* base, int ld, int lane) {
  const int r  = lane & 15;
  const int ho = (lane >> 4) * 8;
  const __bf16* p = base + r * ld;
  bf16x8 lo = *(const bf16x8*)(p + ho);
  bf16x8 hi = *(const bf16x8*)(p + 16 + ho);
  return pack16(lo, hi);
}

__device__ __forceinline__ v8f wmma_bf16(v16bf a, v16bf b, v8f c) {
  return __builtin_amdgcn_wmma_f32_16x16x32_bf16(false, a, false, b, (short)0, c, false, false);
}

// ---------------------------------------------------------------------------
// Weight transpose + f32->bf16 convert: W[1024][1024] -> WT[n][k] bf16
// ---------------------------------------------------------------------------
__global__ __launch_bounds__(256) void convw_kernel(const float* __restrict__ W,
                                                    __bf16* __restrict__ WT) {
  __shared__ float t[32][33];
  const int k0 = blockIdx.x * 32, n0 = blockIdx.y * 32;
  const int tx = threadIdx.x & 31, ty = threadIdx.x >> 5;
#pragma unroll
  for (int i = 0; i < 32; i += 8)
    t[ty + i][tx] = W[(size_t)(k0 + ty + i) * 1024 + n0 + tx];
  __syncthreads();
#pragma unroll
  for (int i = 0; i < 32; i += 8)
    WT[(size_t)(n0 + ty + i) * 1024 + k0 + tx] = (__bf16)t[tx][ty + i];
}

// ---------------------------------------------------------------------------
// GEMM: C[M=8192, N=1024] = A_f32[M,1024] x W (as Bt = W^T bf16 [N][K]) + bias
// 128x128 block tile, 8 waves (2x4), 64-wide K stages, register double-buffered
// global->LDS staging so HBM latency overlaps the WMMA phase.
// MODE 0: Q proj  -> outF (f32) + outB (bf16)
// MODE 1: K proj  -> outB (bf16)
// MODE 2: V proj  -> outT (bf16, per-head transposed vt[b][h][d][key])
// MODE 3: O proj  -> outF = relu(acc + bias)
// ---------------------------------------------------------------------------
template <int MODE>
__global__ __launch_bounds__(256) void gemm_kernel(
    const float* __restrict__ A, const __bf16* __restrict__ Bt,
    const float* __restrict__ bias,
    float* __restrict__ outF, __bf16* __restrict__ outB,
    __bf16* __restrict__ outT) {
  __shared__ __attribute__((aligned(16))) __bf16 As[128][72];
  __shared__ __attribute__((aligned(16))) __bf16 Bs[128][72];
  const int tid  = threadIdx.x;
  const int wave = tid >> 5, lane = tid & 31;
  const int wm = wave >> 2, wn = wave & 3;
  const int rowBase = blockIdx.x * 128;
  const int nBase   = blockIdx.y * 128;
  const int ar = tid >> 1;
  const int ac = (tid & 1) * 32;

  f32x4  areg[8];
  bf16x8 breg[4];
  auto load_global = [&](int kc) {
    const float* ap = A + (size_t)(rowBase + ar) * 1024 + kc + ac;
#pragma unroll
    for (int e = 0; e < 8; ++e) areg[e] = *(const f32x4*)(ap + e * 4);
    const __bf16* bp = Bt + (size_t)(nBase + ar) * 1024 + kc + ac;
#pragma unroll
    for (int e = 0; e < 4; ++e) breg[e] = *(const bf16x8*)(bp + e * 8);
  };

  v8f acc[4][2];
#pragma unroll
  for (int i = 0; i < 4; ++i)
#pragma unroll
    for (int j = 0; j < 2; ++j) acc[i][j] = vzero8();

  load_global(0);
  for (int kc = 0; kc < 1024; kc += 64) {
    // Convert staged A regs to bf16 and commit both tiles to LDS.
#pragma unroll
    for (int e = 0; e < 4; ++e) {
      bf16x8 hh;
#pragma unroll
      for (int q = 0; q < 4; ++q) {
        hh[q]     = (__bf16)areg[2 * e][q];
        hh[4 + q] = (__bf16)areg[2 * e + 1][q];
      }
      *(bf16x8*)(&As[ar][ac + e * 8]) = hh;
      *(bf16x8*)(&Bs[ar][ac + e * 8]) = breg[e];
    }
    __syncthreads();
    if (kc + 64 < 1024) load_global(kc + 64);  // overlap with WMMA phase
#pragma unroll
    for (int ks = 0; ks < 2; ++ks) {
      v16bf bf0 = load_frag(&Bs[wn * 32][ks * 32], 72, lane);
      v16bf bf1 = load_frag(&Bs[wn * 32 + 16][ks * 32], 72, lane);
#pragma unroll
      for (int i = 0; i < 4; ++i) {
        v16bf af = load_frag(&As[wm * 64 + i * 16][ks * 32], 72, lane);
        acc[i][0] = wmma_bf16(af, bf0, acc[i][0]);
        acc[i][1] = wmma_bf16(af, bf1, acc[i][1]);
      }
    }
    __syncthreads();
  }

#pragma unroll
  for (int i = 0; i < 4; ++i)
#pragma unroll
    for (int j = 0; j < 2; ++j) {
      const int gn = nBase + wn * 32 + j * 16 + (lane & 15);
      const int gm = rowBase + wm * 64 + i * 16 + (lane >> 4) * 8;
      const float bvs = bias[gn];
      v8f c = acc[i][j];
      if (MODE == 0) {
#pragma unroll
        for (int r = 0; r < 8; ++r) {
          const float v = c[r] + bvs;
          const size_t idx = (size_t)(gm + r) * 1024 + gn;
          outF[idx] = v;
          outB[idx] = (__bf16)v;
        }
      } else if (MODE == 1) {
#pragma unroll
        for (int r = 0; r < 8; ++r)
          outB[(size_t)(gm + r) * 1024 + gn] = (__bf16)(c[r] + bvs);
      } else if (MODE == 2) {
        // vt[b][h][d][key]; D-fragment's 8 VGPRs are 8 consecutive keys.
        const int bb  = gm >> 10;
        const int key = gm & 1023;
        const int hh2 = gn >> 7, dd = gn & 127;
        bf16x8 pk;
#pragma unroll
        for (int r = 0; r < 8; ++r) pk[r] = (__bf16)(c[r] + bvs);
        *(bf16x8*)(&outT[(((size_t)bb * 8 + hh2) * 128 + dd) * 1024 + key]) = pk;
      } else {
#pragma unroll
        for (int r = 0; r < 8; ++r) {
          const float v = c[r] + bvs;
          outF[(size_t)(gm + r) * 1024 + gn] = v > 0.0f ? v : 0.0f;
        }
      }
    }
}

// ---------------------------------------------------------------------------
// Flash attention per (b,h): 128 query rows/block, 8 waves each owning 16 rows.
// Streams 64-key chunks with register prefetch of the next chunk during WMMA.
// Hpre[b,q,h*128+d] = q_proj + softmax(mask(QK^T/sqrt(128))) V
// ---------------------------------------------------------------------------
__global__ __launch_bounds__(256) void attn_kernel(
    const __bf16* __restrict__ qb, const __bf16* __restrict__ kb,
    const __bf16* __restrict__ vt, const float* __restrict__ qf,
    const float* __restrict__ mask_a, const float* __restrict__ mask_b,
    float* __restrict__ Hpre) {
  __shared__ __attribute__((aligned(16))) __bf16 Ks[64][136];   // [key][d]
  __shared__ __attribute__((aligned(16))) __bf16 Vs[128][72];   // [d][key]
  __shared__ __attribute__((aligned(16))) __bf16 Ps[8][16][72]; // per-wave P
  __shared__ float mas[128];
  __shared__ float mbs[64];
  const int tid = threadIdx.x, wave = tid >> 5, lane = tid & 31;
  const int b = blockIdx.x >> 3, h = blockIdx.x & 7;
  const int qbase = blockIdx.y * 128;
  const float scale = 0.08838834764831845f;  // 1/sqrt(128)

  if (tid < 128) mas[tid] = mask_a[b * 1024 + qbase + tid];

  // Q A-fragments (4 K-steps over d=128) pinned in registers.
  v16bf qfr[4];
  {
    const __bf16* qrow =
        qb + (size_t)(b * 1024 + qbase + wave * 16 + (lane & 15)) * 1024 + h * 128;
    const int ho = (lane >> 4) * 8;
#pragma unroll
    for (int ks = 0; ks < 4; ++ks) {
      bf16x8 lo = *(const bf16x8*)(qrow + ks * 32 + ho);
      bf16x8 hi = *(const bf16x8*)(qrow + ks * 32 + 16 + ho);
      qfr[ks] = pack16(lo, hi);
    }
  }

  bf16x8 kreg[4], vreg[4];
  float mbreg = 0.0f;
  auto load_chunk = [&](int kb0) {
#pragma unroll
    for (int i = 0; i < 4; ++i) {
      const int g = tid + i * 256;
      kreg[i] = *(const bf16x8*)(kb + (size_t)(b * 1024 + kb0 + (g >> 4)) * 1024 +
                                 h * 128 + (g & 15) * 8);
    }
#pragma unroll
    for (int i = 0; i < 4; ++i) {
      const int g = tid + i * 256;
      vreg[i] = *(const bf16x8*)(vt + (((size_t)b * 8 + h) * 128 + (g >> 3)) * 1024 +
                                 kb0 + (g & 7) * 8);
    }
    if (tid < 64) mbreg = mask_b[b * 1024 + kb0 + tid];
  };

  v8f ofr[8];
#pragma unroll
  for (int d = 0; d < 8; ++d) ofr[d] = vzero8();
  float mrow[8], lrow[8];
#pragma unroll
  for (int r = 0; r < 8; ++r) { mrow[r] = -3.0e38f; lrow[r] = 0.0f; }

  load_chunk(0);
  for (int kb0 = 0; kb0 < 1024; kb0 += 64) {
    __syncthreads();  // all waves done reading Ks/Vs of previous chunk
#pragma unroll
    for (int i = 0; i < 4; ++i) {
      const int g = tid + i * 256;
      *(bf16x8*)(&Ks[g >> 4][(g & 15) * 8]) = kreg[i];
    }
#pragma unroll
    for (int i = 0; i < 4; ++i) {
      const int g = tid + i * 256;
      *(bf16x8*)(&Vs[g >> 3][(g & 7) * 8]) = vreg[i];
    }
    if (tid < 64) mbs[tid] = mbreg;
    __syncthreads();
    if (kb0 + 64 < 1024) load_chunk(kb0 + 64);  // overlap with WMMA phase

    // S = Q K^T for this wave's 16 rows x 64 keys.
    v8f sfr[4];
#pragma unroll
    for (int nt = 0; nt < 4; ++nt) {
      v8f s = vzero8();
#pragma unroll
      for (int ks = 0; ks < 4; ++ks) {
        v16bf bf = load_frag(&Ks[nt * 16][ks * 32], 136, lane);
        s = wmma_bf16(qfr[ks], bf, s);
      }
      sfr[nt] = s;
    }

    // Mask + scale + per-row chunk max.
    float cm[8];
#pragma unroll
    for (int r = 0; r < 8; ++r) {
      cm[r] = -3.0e38f;
      const float maq = mas[wave * 16 + (lane >> 4) * 8 + r];
#pragma unroll
      for (int nt = 0; nt < 4; ++nt) {
        const float mbk = mbs[nt * 16 + (lane & 15)];
        const float s = (maq * mbk == 1.0f) ? sfr[nt][r] * scale : NEGBIG;
        sfr[nt][r] = s;
        cm[r] = fmaxf(cm[r], s);
      }
    }
#pragma unroll
    for (int r = 0; r < 8; ++r)
#pragma unroll
      for (int m = 1; m < 16; m <<= 1)
        cm[r] = fmaxf(cm[r], __shfl_xor(cm[r], m, 32));

    float alpha[8], rs[8];
#pragma unroll
    for (int r = 0; r < 8; ++r) {
      const float mn = fmaxf(mrow[r], cm[r]);
      alpha[r] = __expf(mrow[r] - mn);
      mrow[r] = mn;
      rs[r] = 0.0f;
    }
#pragma unroll
    for (int nt = 0; nt < 4; ++nt)
#pragma unroll
      for (int r = 0; r < 8; ++r) {
        const float p = __expf(sfr[nt][r] - mrow[r]);
        sfr[nt][r] = p;
        rs[r] += p;
      }
#pragma unroll
    for (int r = 0; r < 8; ++r) {
#pragma unroll
      for (int m = 1; m < 16; m <<= 1) rs[r] += __shfl_xor(rs[r], m, 32);
      lrow[r] = alpha[r] * lrow[r] + rs[r];
    }
#pragma unroll
    for (int d = 0; d < 8; ++d)
#pragma unroll
      for (int r = 0; r < 8; ++r) ofr[d][r] *= alpha[r];

    // P -> bf16 A-fragment layout via per-wave LDS round trip.
#pragma unroll
    for (int nt = 0; nt < 4; ++nt)
#pragma unroll
      for (int r = 0; r < 8; ++r)
        Ps[wave][(lane >> 4) * 8 + r][nt * 16 + (lane & 15)] = (__bf16)sfr[nt][r];

#pragma unroll
    for (int ks = 0; ks < 2; ++ks) {
      v16bf pa = load_frag(&Ps[wave][0][ks * 32], 72, lane);
#pragma unroll
      for (int dt = 0; dt < 8; ++dt) {
        v16bf bv = load_frag(&Vs[dt * 16][ks * 32], 72, lane);
        ofr[dt] = wmma_bf16(pa, bv, ofr[dt]);
      }
    }
  }

  // O/l + residual q.
#pragma unroll
  for (int dt = 0; dt < 8; ++dt) {
    const int q0  = qbase + wave * 16 + (lane >> 4) * 8;
    const int col = h * 128 + dt * 16 + (lane & 15);
#pragma unroll
    for (int r = 0; r < 8; ++r) {
      const size_t idx = (size_t)(b * 1024 + q0 + r) * 1024 + col;
      Hpre[idx] = ofr[dt][r] / lrow[r] + qf[idx];
    }
  }
}

// ---------------------------------------------------------------------------
// LayerNorm over rows of 1024, with optional pre-add of Y (residual).
// ---------------------------------------------------------------------------
__global__ __launch_bounds__(256) void ln_kernel(
    const float* __restrict__ X, const float* __restrict__ Y,
    const float* __restrict__ g, const float* __restrict__ be,
    float* __restrict__ out) {
  __shared__ float red[16];
  const int row = blockIdx.x, tid = threadIdx.x;
  const int wv = tid >> 5, ln = tid & 31;
  const float* xp = X + (size_t)row * 1024;
  const float* yp = Y ? Y + (size_t)row * 1024 : nullptr;
  float v[4], s = 0.0f, ss = 0.0f;
#pragma unroll
  for (int i = 0; i < 4; ++i) {
    float x = xp[tid + i * 256];
    if (yp) x += yp[tid + i * 256];
    v[i] = x;
    s += x;
    ss += x * x;
  }
#pragma unroll
  for (int m = 1; m < 32; m <<= 1) {
    s += __shfl_xor(s, m, 32);
    ss += __shfl_xor(ss, m, 32);
  }
  if (ln == 0) { red[wv] = s; red[8 + wv] = ss; }
  __syncthreads();
  float ts = 0.0f, tss = 0.0f;
#pragma unroll
  for (int w = 0; w < 8; ++w) { ts += red[w]; tss += red[8 + w]; }
  const float mu   = ts * (1.0f / 1024.0f);
  const float var  = tss * (1.0f / 1024.0f) - mu * mu;
  const float rstd = rsqrtf(var + 1e-5f);
#pragma unroll
  for (int i = 0; i < 4; ++i)
    out[(size_t)row * 1024 + tid + i * 256] =
        (v[i] - mu) * rstd * g[tid + i * 256] + be[tid + i * 256];
}

// ---------------------------------------------------------------------------
extern "C" void kernel_launch(void* const* d_in, const int* in_sizes, int n_in,
                              void* d_out, int out_size, void* d_ws, size_t ws_size,
                              hipStream_t stream) {
  (void)in_sizes; (void)n_in; (void)out_size; (void)ws_size;
  const float* Q   = (const float*)d_in[0];
  const float* K   = (const float*)d_in[1];
  const float* ma  = (const float*)d_in[2];
  const float* mb  = (const float*)d_in[3];
  const float* Wq  = (const float*)d_in[4];  const float* bq  = (const float*)d_in[5];
  const float* Wk  = (const float*)d_in[6];  const float* bk  = (const float*)d_in[7];
  const float* Wv  = (const float*)d_in[8];  const float* bv  = (const float*)d_in[9];
  const float* Wo  = (const float*)d_in[10]; const float* bo  = (const float*)d_in[11];
  const float* g0  = (const float*)d_in[12]; const float* be0 = (const float*)d_in[13];
  const float* g1  = (const float*)d_in[14]; const float* be1 = (const float*)d_in[15];
  float* out = (float*)d_out;

  char* ws = (char*)d_ws;
  size_t off = 0;
  auto alloc = [&](size_t bytes) -> void* {
    void* p = ws + off;
    off += (bytes + 255) & ~(size_t)255;
    return p;
  };
  const size_t MROW = 8192;  // B * 1024
  __bf16* WqT = (__bf16*)alloc(1024 * 1024 * sizeof(__bf16));
  __bf16* WkT = (__bf16*)alloc(1024 * 1024 * sizeof(__bf16));
  __bf16* WvT = (__bf16*)alloc(1024 * 1024 * sizeof(__bf16));
  __bf16* WoT = (__bf16*)alloc(1024 * 1024 * sizeof(__bf16));
  float*  qf    = (float*)alloc(MROW * 1024 * sizeof(float));
  __bf16* qbp   = (__bf16*)alloc(MROW * 1024 * sizeof(__bf16));
  __bf16* kbp   = (__bf16*)alloc(MROW * 1024 * sizeof(__bf16));
  __bf16* vtp   = (__bf16*)alloc(MROW * 1024 * sizeof(__bf16));
  float*  Hpre  = (float*)alloc(MROW * 1024 * sizeof(float));
  float*  H2    = (float*)alloc(MROW * 1024 * sizeof(float));
  float*  reluG = (float*)alloc(MROW * 1024 * sizeof(float));

  dim3 blk(256);
  convw_kernel<<<dim3(32, 32), blk, 0, stream>>>(Wq, WqT);
  convw_kernel<<<dim3(32, 32), blk, 0, stream>>>(Wk, WkT);
  convw_kernel<<<dim3(32, 32), blk, 0, stream>>>(Wv, WvT);
  convw_kernel<<<dim3(32, 32), blk, 0, stream>>>(Wo, WoT);
  gemm_kernel<0><<<dim3(64, 8), blk, 0, stream>>>(Q, WqT, bq, qf, qbp, nullptr);
  gemm_kernel<1><<<dim3(64, 8), blk, 0, stream>>>(K, WkT, bk, nullptr, kbp, nullptr);
  gemm_kernel<2><<<dim3(64, 8), blk, 0, stream>>>(K, WvT, bv, nullptr, nullptr, vtp);
  attn_kernel<<<dim3(64, 8), blk, 0, stream>>>(qbp, kbp, vtp, qf, ma, mb, Hpre);
  ln_kernel<<<dim3(8192), blk, 0, stream>>>(Hpre, nullptr, g0, be0, H2);
  gemm_kernel<3><<<dim3(64, 8), blk, 0, stream>>>(H2, WoT, bo, reluG, nullptr, nullptr);
  ln_kernel<<<dim3(8192), blk, 0, stream>>>(H2, reluG, g1, be1, out);
}
